// SAFM11_41266045780174
// MI455X (gfx1250) — compile-verified, matrix operand order checked by reading, not run on hardware
//
#include <hip/hip_runtime.h>
#include <hip/hip_bf16.h>
#include <math.h>

typedef __attribute__((ext_vector_type(16))) _Float16 v16h;
typedef __attribute__((ext_vector_type(8)))  _Float16 v8h;
typedef __attribute__((ext_vector_type(8)))  float    v8f;

#define BATCH 8
#define CH    64
#define IMG_H 256
#define IMG_W 256
#define HW    (IMG_H*IMG_W)

#define H_TILE 8
#define W_TILE 32
#define TP     (H_TILE*W_TILE)   // 256 pixels per block
#define CATP   72                // padded cat row (f16): 36 dwords -> conflict-free banks

// load 16 f16 as two 16B-aligned halves (rows are 144B = 16B aligned only)
__device__ __forceinline__ v16h load16h(const _Float16* p) {
    union { v16h v; v8h h[2]; } u;
    u.h[0] = *(const v8h*)p;
    u.h[1] = *(const v8h*)(p + 8);
    return u.v;
}

// ---------- small pyramid precompute kernels (tiny traffic) ----------------

__global__ void pool4_kernel(const float* __restrict__ src, float* __restrict__ dst,
                             int Csrc, int coff, int Cdst, int Hout, int Wout) {
    const int N   = BATCH * Cdst * Hout * Wout;
    const int Hin = Hout * 4, Win = Wout * 4;
    for (int idx = blockIdx.x * blockDim.x + threadIdx.x; idx < N;
         idx += gridDim.x * blockDim.x) {
        int wo = idx % Wout;
        int ho = (idx / Wout) % Hout;
        int c  = (idx / (Wout * Hout)) % Cdst;
        int b  = idx / (Wout * Hout * Cdst);
        const float* s = src + (((size_t)b * Csrc + coff + c) * Hin + (size_t)ho * 4) * Win + wo * 4;
        float m = -INFINITY;
        for (int dy = 0; dy < 4; ++dy)
            for (int dx = 0; dx < 4; ++dx)
                m = fmaxf(m, s[dy * Win + dx]);
        dst[idx] = m;
    }
}

__global__ void dwconv3_kernel(const float* __restrict__ src, const float* __restrict__ wgt,
                               const float* __restrict__ bias, float* __restrict__ dst,
                               int Csrc, int coff, int Cdst, int Hh, int Ww) {
    const int N = BATCH * Cdst * Hh * Ww;
    for (int idx = blockIdx.x * blockDim.x + threadIdx.x; idx < N;
         idx += gridDim.x * blockDim.x) {
        int w = idx % Ww;
        int h = (idx / Ww) % Hh;
        int c = (idx / (Ww * Hh)) % Cdst;
        int b = idx / (Ww * Hh * Cdst);
        const float* s = src + ((size_t)b * Csrc + coff + c) * (size_t)(Hh * Ww);
        float acc = bias[c];
        for (int dh = -1; dh <= 1; ++dh) {
            int hh = h + dh;
            if (hh < 0 || hh >= Hh) continue;
            for (int dw = -1; dw <= 1; ++dw) {
                int ww = w + dw;
                if (ww < 0 || ww >= Ww) continue;
                acc += wgt[c * 9 + (dh + 1) * 3 + (dw + 1)] * s[hh * Ww + ww];
            }
        }
        dst[idx] = acc;
    }
}

__global__ void cvt_wa_kernel(const float* __restrict__ wa, _Float16* __restrict__ wa_h) {
    int i = blockIdx.x * blockDim.x + threadIdx.x;
    if (i < 64 * 64) wa_h[i] = (_Float16)wa[i];
}

// ---------- fused full-res kernel: dwconv + gather + WMMA mix + GELU*x -----

__global__ __launch_bounds__(256) void safm_fused_kernel(
    const float* __restrict__ x,  const float* __restrict__ w1, const float* __restrict__ b1,
    const _Float16* __restrict__ wa_h, const float* __restrict__ ba,
    const float* __restrict__ x2c, const float* __restrict__ x3c, const float* __restrict__ x4c,
    float* __restrict__ out)
{
    __shared__ _Float16 cat_s[TP][CATP];                 // B operand, [pixel][c], padded
    __shared__ _Float16 xa_s[32][H_TILE + 2][W_TILE + 4]; // conv input tile + halo (f16)
    __shared__ float w1_s[32 * 9];
    __shared__ float b1_s[32];
    __shared__ float ba_s[64];

    const int tid = threadIdx.x;
    const int b   = blockIdx.z;
    const int h0  = blockIdx.y * H_TILE;
    const int w0  = blockIdx.x * W_TILE;

    // ---- prefetch the x channels (32..63) needed only by the final multiply
    {
        int c = 32 + (tid >> 3);     // 32 channels x 8 rows = 256 cachelines
        int r = tid & 7;
        const float* pf = &x[((size_t)b * CH + c) * (size_t)HW + (size_t)(h0 + r) * IMG_W + w0];
        __builtin_prefetch(pf, 0, 0);
    }

    // ---- stage 0: small weights into LDS ----------------------------------
    for (int i = tid; i < 32 * 9; i += 256) w1_s[i] = w1[i];
    if (tid < 32) b1_s[tid] = b1[tid];
    if (tid < 64) ba_s[tid] = ba[tid];

    // ---- stage 1: xa tile with 1-pixel halo (f32 -> f16) ------------------
    for (int i = tid; i < 32 * (H_TILE + 2) * (W_TILE + 2); i += 256) {
        int c  = i / ((H_TILE + 2) * (W_TILE + 2));
        int r  = (i / (W_TILE + 2)) % (H_TILE + 2);
        int cc = i % (W_TILE + 2);
        int h = h0 + r - 1;
        int w = w0 + cc - 1;
        float v = 0.0f;
        if (h >= 0 && h < IMG_H && w >= 0 && w < IMG_W)
            v = x[((size_t)b * CH + c) * (size_t)HW + (size_t)h * IMG_W + w];
        xa_s[c][r][cc] = (_Float16)v;
    }
    __syncthreads();

    // ---- stage 2: build cat[pixel][64] in f16 (B-operand layout) ----------
    {
        int p  = tid;                // one pixel per thread, no divergence
        int r  = (p >> 5) + 1;       // row in halo coords
        int cc = (p & 31) + 1;       // col in halo coords
        for (int c = 0; c < 32; ++c) {
            const float* wk = &w1_s[c * 9];
            float acc = b1_s[c];
            acc += wk[0]*(float)xa_s[c][r-1][cc-1] + wk[1]*(float)xa_s[c][r-1][cc] + wk[2]*(float)xa_s[c][r-1][cc+1];
            acc += wk[3]*(float)xa_s[c][r  ][cc-1] + wk[4]*(float)xa_s[c][r  ][cc] + wk[5]*(float)xa_s[c][r  ][cc+1];
            acc += wk[6]*(float)xa_s[c][r+1][cc-1] + wk[7]*(float)xa_s[c][r+1][cc] + wk[8]*(float)xa_s[c][r+1][cc+1];
            cat_s[p][c] = (_Float16)acc;
        }
        int h = h0 + (p >> 5);
        int w = w0 + (p & 31);
        const float* s2 = x2c + (size_t)b * 16 * 64 * 64;   // nearest-up == index div
        for (int i = 0; i < 16; ++i)
            cat_s[p][32 + i] = (_Float16)s2[((size_t)i * 64 + (h >> 2)) * 64 + (w >> 2)];
        const float* s3 = x3c + (size_t)b * 8 * 16 * 16;
        for (int i = 0; i < 8; ++i)
            cat_s[p][48 + i] = (_Float16)s3[((size_t)i * 16 + (h >> 4)) * 16 + (w >> 4)];
        const float* s4 = x4c + (size_t)b * 8 * 4 * 4;
        for (int i = 0; i < 8; ++i)
            cat_s[p][56 + i] = (_Float16)s4[((size_t)i * 4 + (h >> 6)) * 4 + (w >> 6)];
    }
    __syncthreads();

    // ---- stage 3: 64x64 channel mix via WMMA, then GELU * x ---------------
    const int lane  = tid & 31;
    const int wv    = tid >> 5;        // wave id == tile row; owns 2 pixel groups
    const int nloc  = lane & 15;       // N (pixel) / M (row) local index
    const int khalf = lane >> 4;       // which 16-wide K slice this lane holds
    const int h     = h0 + wv;

    for (int gsub = 0; gsub < 2; ++gsub) {
        const int colb = gsub * 16;
        const int p = wv * 32 + colb + nloc;
        // B operand: lane holds K-slice [khalf*16 .. +15] of column (pixel) nloc
        v16h bm0 = load16h(&cat_s[p][khalf * 16]);
        v16h bm1 = load16h(&cat_s[p][32 + khalf * 16]);
        const int w = w0 + colb + nloc;

        for (int t = 0; t < 4; ++t) {  // 4 tiles of 16 output channels
            v8f acc = {};
            v16h a0 = load16h(&wa_h[(t * 16 + nloc) * 64 + khalf * 16]);
            acc = __builtin_amdgcn_wmma_f32_16x16x32_f16(false, a0, false, bm0,
                                                         (short)0, acc, false, false);
            v16h a1 = load16h(&wa_h[(t * 16 + nloc) * 64 + 32 + khalf * 16]);
            acc = __builtin_amdgcn_wmma_f32_16x16x32_f16(false, a1, false, bm1,
                                                         (short)0, acc, false, false);
            // D layout: vgpr j, lane -> channel t*16 + khalf*8 + j, pixel nloc
            for (int j = 0; j < 8; ++j) {
                int o = t * 16 + khalf * 8 + j;
                float v = acc[j] + ba_s[o];
                float g = 0.5f * v * (1.0f + erff(v * 0.70710678118654752f));
                size_t idx = ((size_t)b * CH + o) * (size_t)HW + (size_t)h * IMG_W + w;
                out[idx] = g * x[idx];
            }
        }
    }
}

// ---------------------------------------------------------------------------

extern "C" void kernel_launch(void* const* d_in, const int* in_sizes, int n_in,
                              void* d_out, int out_size, void* d_ws, size_t ws_size,
                              hipStream_t stream) {
    (void)in_sizes; (void)n_in; (void)out_size; (void)ws_size;
    const float* x  = (const float*)d_in[0];
    const float* w1 = (const float*)d_in[1];
    const float* b1 = (const float*)d_in[2];
    const float* w2 = (const float*)d_in[3];
    const float* b2 = (const float*)d_in[4];
    const float* w3 = (const float*)d_in[5];
    const float* b3 = (const float*)d_in[6];
    const float* w4 = (const float*)d_in[7];
    const float* b4 = (const float*)d_in[8];
    const float* wa = (const float*)d_in[9];
    const float* ba = (const float*)d_in[10];
    float* out = (float*)d_out;

    char* ws = (char*)d_ws;
    float*     p2   = (float*)(ws);                                // 8*32*64*64 f32 = 4 MB
    float*     x2c  = (float*)(ws + 4u*1024*1024);                 // 2 MB
    float*     p3   = (float*)(ws + 6u*1024*1024);                 // 128 KB
    float*     x3c  = (float*)(ws + 6u*1024*1024 + 256u*1024);     // 64 KB
    float*     p4   = (float*)(ws + 6u*1024*1024 + 512u*1024);     // 4 KB
    float*     x4c  = (float*)(ws + 6u*1024*1024 + 768u*1024);     // 4 KB
    _Float16*  wa_h = (_Float16*)(ws + 7u*1024*1024);              // 8 KB, f16 mix matrix

    // pyramid precompute (all tiny except p2 which reads xb once)
    { int N = BATCH*32*64*64; pool4_kernel  <<<(N+255)/256, 256, 0, stream>>>(x,  p2,  64, 32, 32, 64, 64); }
    { int N = BATCH*16*64*64; dwconv3_kernel<<<(N+255)/256, 256, 0, stream>>>(p2, w2, b2, x2c, 32, 0, 16, 64, 64); }
    { int N = BATCH*16*16*16; pool4_kernel  <<<(N+255)/256, 256, 0, stream>>>(p2, p3,  32, 16, 16, 16, 16); }
    { int N = BATCH*8*16*16;  dwconv3_kernel<<<(N+255)/256, 256, 0, stream>>>(p3, w3, b3, x3c, 16, 0, 8, 16, 16); }
    { int N = BATCH*8*4*4;    pool4_kernel  <<<(N+255)/256, 256, 0, stream>>>(p3, p4,  16,  8,  8,  4,  4); }
    { int N = BATCH*8*4*4;    dwconv3_kernel<<<(N+255)/256, 256, 0, stream>>>(p4, w4, b4, x4c, 8, 0, 8, 4, 4); }
    cvt_wa_kernel<<<16, 256, 0, stream>>>(wa, wa_h);

    // fused full-resolution pass
    dim3 grid(IMG_W / W_TILE, IMG_H / H_TILE, BATCH);
    safm_fused_kernel<<<grid, 256, 0, stream>>>(x, w1, b1, wa_h, ba, x2c, x3c, x4c, out);
}
